// MultiHeadAttn_56307021250631
// MI455X (gfx1250) — matrix operand order repaired and checked by reading, not compiled, not run on hardware
//
#include <hip/hip_runtime.h>
#include <hip/hip_bf16.h>

// ---------------------------------------------------------------------------
// MI455X (gfx1250) multi-head attention, all matmuls on v_wmma_f32_16x16x32_f16.
// Every GEMM is arranged "NT" (contraction dim contiguous in BOTH operands) so
// A- and B-fragments are plain 16B contiguous loads matching the CDNA5 WMMA
// VGPR layouts (no transpose loads needed). V projection is stored transposed
// [B,H,HD,S] so ctx = P @ V is also NT. Attention scores (4MB) + P (2MB) are
// reused per (b,h) so the softmax working set lives in the 192MB L2.
// Big GEMMs (K=1024) use 64x64 wave tiles (16 WMMA / 16 b128 loads per k-step,
// ~32 flops/byte of operand traffic); the N=64 ctx GEMM uses 32x32 wave tiles.
// ---------------------------------------------------------------------------

typedef __attribute__((ext_vector_type(16))) _Float16 v16h;
typedef __attribute__((ext_vector_type(8)))  float    v8f;
typedef __attribute__((ext_vector_type(4)))  float    f32x4;
typedef __attribute__((ext_vector_type(4)))  _Float16 f16x4;

// ---- fragment loaders -------------------------------------------------------
// A-matrix 16x32 f16 layout (ISA 7.12.2): lane L holds row M=L%16;
//   v0..v3 = K[(L/16)*8 .. +7], v4..v7 = K[(L/16)*8+16 .. +23]
__device__ __forceinline__ v16h load_a_frag(const _Float16* __restrict__ base,
                                            int ld, int lane) {
  const _Float16* p = base + (size_t)(lane & 15) * ld + ((lane >> 4) << 3);
  union { v16h h; f32x4 f[2]; } u;
  u.f[0] = *reinterpret_cast<const f32x4*>(p);
  u.f[1] = *reinterpret_cast<const f32x4*>(p + 16);
  return u.h;
}

// B-matrix 32x16 f16 layout: lane L holds column N=L%16;
//   v0..v3 = K[(L/16)*8 .. +7], v4..v7 = K[(L/16)*8+16 .. +23] (same as A)
// Our "B" pointer is row-major [N,K] (weight rows / K rows / V^T rows), so a
// column of the math-B is a contiguous row in memory.
__device__ __forceinline__ v16h load_b_frag(const _Float16* __restrict__ base,
                                            int ld, int lane) {
  const _Float16* p = base + (size_t)(lane & 15) * ld + ((lane >> 4) << 3);
  union { v16h h; f32x4 f[2]; } u;
  u.f[0] = *reinterpret_cast<const f32x4*>(p);
  u.f[1] = *reinterpret_cast<const f32x4*>(p + 16);
  return u.h;
}

__device__ __forceinline__ v8f wmma_f16(v16h a, v16h b, v8f c) {
  return __builtin_amdgcn_wmma_f32_16x16x32_f16(
      /*neg_a=*/false, a, /*neg_b=*/false, b,
      /*c_mod=*/(short)0, c, /*reuse_a=*/false, /*reuse_b=*/false);
}

// ---- GEMM: C[M,N] = alpha * A[M,K] * B[N,K]^T -------------------------------
// MODE 0: C f32 row-major (ldc)
// MODE 1: C f16 row-major (ldc)
// MODE 2: C f16 scattered to V^T layout [B=4,H=16,HD=64,S=1024]
//         (row gm in [0,B*S) -> b,s ; col gn in [0,E) -> h,d)
// Block = 256 threads = 8 waves arranged WROWS x WCOLS; each wave computes a
// (MI*16) x (NJ*16) tile of C (MI x NJ WMMA accumulators).
// All problem dims divide the tiles exactly -> no divergence, EXEC all ones.
template <int MODE, int MI, int NJ, int WROWS, int WCOLS>
__global__ __launch_bounds__(256) void gemm_nt(
    const _Float16* __restrict__ A, const _Float16* __restrict__ B,
    void* __restrict__ C, int K, int lda, int ldb, int ldc, float alpha) {
  static_assert(WROWS * WCOLS == 8, "8 waves per block");
  const int lane = threadIdx.x & 31;
  const int wid  = threadIdx.x >> 5;  // 0..7
  const int m0 = blockIdx.y * (WROWS * MI * 16) + (wid / WCOLS) * (MI * 16);
  const int n0 = blockIdx.x * (WCOLS * NJ * 16) + (wid % WCOLS) * (NJ * 16);

  v8f acc[MI][NJ] = {};

  const _Float16* Arow[MI];
  const _Float16* Brow[NJ];
#pragma unroll
  for (int i = 0; i < MI; ++i) Arow[i] = A + (size_t)(m0 + i * 16) * lda;
#pragma unroll
  for (int j = 0; j < NJ; ++j) Brow[j] = B + (size_t)(n0 + j * 16) * ldb;

  for (int k0 = 0; k0 < K; k0 += 32) {
    if (k0 + 32 < K) {  // hint next k-tile into cache (global_prefetch_b8)
      __builtin_prefetch(Arow[0] + k0 + 32, 0, 3);
      __builtin_prefetch(Brow[0] + k0 + 32, 0, 3);
    }
    v16h a[MI], b[NJ];
#pragma unroll
    for (int i = 0; i < MI; ++i) a[i] = load_a_frag(Arow[i] + k0, lda, lane);
#pragma unroll
    for (int j = 0; j < NJ; ++j) b[j] = load_b_frag(Brow[j] + k0, ldb, lane);
#pragma unroll
    for (int i = 0; i < MI; ++i)
#pragma unroll
      for (int j = 0; j < NJ; ++j)
        acc[i][j] = wmma_f16(a[i], b[j], acc[i][j]);
  }

  // C/D 16x16 f32 layout: VGPR r, lane L -> M = r + 8*(L/16), N = L%16
  const int cn = lane & 15;
  const int rh = (lane >> 4) * 8;
#pragma unroll
  for (int ti = 0; ti < MI; ++ti)
#pragma unroll
    for (int tj = 0; tj < NJ; ++tj)
#pragma unroll
      for (int r = 0; r < 8; ++r) {
        const int gm = m0 + ti * 16 + rh + r;
        const int gn = n0 + tj * 16 + cn;
        const float val = acc[ti][tj][r] * alpha;
        if (MODE == 0) {
          reinterpret_cast<float*>(C)[(size_t)gm * ldc + gn] = val;
        } else if (MODE == 1) {
          reinterpret_cast<_Float16*>(C)[(size_t)gm * ldc + gn] = (_Float16)val;
        } else {  // V^T store: [b,h,d,s] with S=1024, H=16, HD=64
          const int bb = gm >> 10, ss = gm & 1023;
          const int hh = gn >> 6,  dd = gn & 63;
          reinterpret_cast<_Float16*>(C)[(((size_t)(bb * 16 + hh) * 64) + dd) * 1024 + ss] =
              (_Float16)val;
        }
      }
}

// ---- row softmax: f32 scores -> f16 probabilities ---------------------------
__global__ __launch_bounds__(256) void softmax_rows(
    const float* __restrict__ S, _Float16* __restrict__ P, int n) {
  __shared__ float red[256];
  const int tid = threadIdx.x;
  const float* src = S + (size_t)blockIdx.x * n;

  float m = -3.4e38f;
  for (int c = tid; c < n; c += 256) m = fmaxf(m, src[c]);
  red[tid] = m;
  __syncthreads();
  for (int s = 128; s > 0; s >>= 1) {
    if (tid < s) red[tid] = fmaxf(red[tid], red[tid + s]);
    __syncthreads();
  }
  m = red[0];
  __syncthreads();

  float sum = 0.f;
  for (int c = tid; c < n; c += 256) sum += __expf(src[c] - m);
  red[tid] = sum;
  __syncthreads();
  for (int s = 128; s > 0; s >>= 1) {
    if (tid < s) red[tid] += red[tid + s];
    __syncthreads();
  }
  const float inv = 1.0f / red[0];

  _Float16* dst = P + (size_t)blockIdx.x * n;
  for (int c = tid; c < n; c += 256)
    dst[c] = (_Float16)(__expf(src[c] - m) * inv);
}

// ---- f32 -> f16 convert (x4 vectorized) -------------------------------------
__global__ __launch_bounds__(256) void cvt_f32_f16(
    const float* __restrict__ x, _Float16* __restrict__ y, int n4) {
  const int i = blockIdx.x * 256 + threadIdx.x;
  if (i < n4) {
    f32x4 v = reinterpret_cast<const f32x4*>(x)[i];
    f16x4 h;
    h.x = (_Float16)v.x; h.y = (_Float16)v.y;
    h.z = (_Float16)v.z; h.w = (_Float16)v.w;
    reinterpret_cast<f16x4*>(y)[i] = h;
  }
}

// ---------------------------------------------------------------------------
extern "C" void kernel_launch(void* const* d_in, const int* in_sizes, int n_in,
                              void* d_out, int out_size, void* d_ws, size_t ws_size,
                              hipStream_t stream) {
  (void)in_sizes; (void)n_in; (void)out_size; (void)ws_size;
  constexpr int Bb = 4, Ss = 1024, Ee = 1024, Hh = 16, HDd = 64;
  constexpr size_t BSE = (size_t)Bb * Ss * Ee;  // 4,194,304
  constexpr size_t EE  = (size_t)Ee * Ee;       // 1,048,576

  const float* q  = (const float*)d_in[0];
  const float* k  = (const float*)d_in[1];
  const float* v  = (const float*)d_in[2];
  const float* Wq = (const float*)d_in[3];
  const float* Wk = (const float*)d_in[4];
  const float* Wv = (const float*)d_in[5];
  const float* Wo = (const float*)d_in[6];
  float* out = (float*)d_out;

  // workspace carve-up (~74 MB)
  char* w = (char*)d_ws;
  _Float16* hq  = (_Float16*)w; w += BSE * 2;
  _Float16* hk  = (_Float16*)w; w += BSE * 2;
  _Float16* hv  = (_Float16*)w; w += BSE * 2;
  _Float16* hWq = (_Float16*)w; w += EE * 2;
  _Float16* hWk = (_Float16*)w; w += EE * 2;
  _Float16* hWv = (_Float16*)w; w += EE * 2;
  _Float16* hWo = (_Float16*)w; w += EE * 2;
  _Float16* Qp  = (_Float16*)w; w += BSE * 2;   // [B,S,E] = [B,S,H,HD]
  _Float16* Kp  = (_Float16*)w; w += BSE * 2;   // [B,S,E]
  _Float16* VT  = (_Float16*)w; w += BSE * 2;   // [B,H,HD,S]
  _Float16* cat = (_Float16*)w; w += BSE * 2;   // [B,S,E]
  float*    sc  = (float*)w;    w += (size_t)Ss * Ss * 4;  // scores, reused per (b,h)
  _Float16* P   = (_Float16*)w; w += (size_t)Ss * Ss * 2;  // probs,  reused per (b,h)

  const dim3 blk(256);

  // 1) convert inputs + weights to f16
  cvt_f32_f16<<<BSE / 4 / 256, blk, 0, stream>>>(q,  hq,  (int)(BSE / 4));
  cvt_f32_f16<<<BSE / 4 / 256, blk, 0, stream>>>(k,  hk,  (int)(BSE / 4));
  cvt_f32_f16<<<BSE / 4 / 256, blk, 0, stream>>>(v,  hv,  (int)(BSE / 4));
  cvt_f32_f16<<<EE  / 4 / 256, blk, 0, stream>>>(Wq, hWq, (int)(EE / 4));
  cvt_f32_f16<<<EE  / 4 / 256, blk, 0, stream>>>(Wk, hWk, (int)(EE / 4));
  cvt_f32_f16<<<EE  / 4 / 256, blk, 0, stream>>>(Wv, hWv, (int)(EE / 4));
  cvt_f32_f16<<<EE  / 4 / 256, blk, 0, stream>>>(Wo, hWo, (int)(EE / 4));

  // Big-tile config: wave 64x64 (4x4 WMMA), waves 2x4 -> block tile 128x256.
  // 16 WMMA per 16 b128 loads per k-step => 2x operand reuse vs 32x32 tiles.
  constexpr int BM = 2 * 4 * 16;  // 128
  constexpr int BN = 4 * 4 * 16;  // 256

  // 2) projections: [B*S,E] x [E,E]^T  (W stored [out,in] row-major -> NT GEMM)
  const dim3 gproj(Ee / BN, (Bb * Ss) / BM);  // (4, 32)
  gemm_nt<1, 4, 4, 2, 4><<<gproj, blk, 0, stream>>>(hq, hWq, Qp, Ee, Ee, Ee, Ee, 1.0f);
  gemm_nt<1, 4, 4, 2, 4><<<gproj, blk, 0, stream>>>(hk, hWk, Kp, Ee, Ee, Ee, Ee, 1.0f);
  gemm_nt<2, 4, 4, 2, 4><<<gproj, blk, 0, stream>>>(hv, hWv, VT, Ee, Ee, Ee, Ee, 1.0f);

  // 3) attention per (b,h): scores -> softmax -> ctx. Scratch reused so the
  //    6 MB working set stays L2-resident; stream ordering serializes reuse.
  const float iscale = 0.125f;                 // 1/sqrt(HD=64)
  const dim3 gsc(Ss / BN, Ss / BM);            // (4, 8)
  const dim3 gctx(HDd / 64, Ss / 128);         // (1, 8), 32x32 wave tiles
  for (int b = 0; b < Bb; ++b) {
    for (int h = 0; h < Hh; ++h) {
      const _Float16* Qbh = Qp + (size_t)b * Ss * Ee + h * HDd;   // lda = E
      const _Float16* Kbh = Kp + (size_t)b * Ss * Ee + h * HDd;   // ldb = E
      gemm_nt<0, 4, 4, 2, 4><<<gsc, blk, 0, stream>>>(Qbh, Kbh, sc,
                                                      HDd, Ee, Ee, Ss, iscale);
      softmax_rows<<<Ss, blk, 0, stream>>>(sc, P, Ss);
      const _Float16* Vbh = VT + (size_t)(b * Hh + h) * HDd * Ss; // rows = V^T
      gemm_nt<1, 2, 2, 4, 2><<<gctx, blk, 0, stream>>>(P, Vbh,
                                                       cat + (size_t)b * Ss * Ee + h * HDd,
                                                       Ss, Ss, Ss, Ee, 1.0f);
    }
  }

  // 4) output projection: [B*S,E] x Wo[E,E]^T -> f32 d_out
  gemm_nt<0, 4, 4, 2, 4><<<gproj, blk, 0, stream>>>(cat, hWo, out, Ee, Ee, Ee, Ee, 1.0f);
}
